// SelfAttention_37864431682006
// MI455X (gfx1250) — matrix operand order repaired and checked by reading, not compiled, not run on hardware
//
#include <hip/hip_runtime.h>
#include <hip/hip_bf16.h>
#include <cmath>

// ---------------------------------------------------------------------------
// Self-attention forward for MI455X (gfx1250, wave32, WMMA + TDM + async-LDS).
//   0) convert x, Wq, Wk, Wv, Wp -> bf16
//   1-3) Q/K = xb*W^T+b (bf16 row-major); V stored TRANSPOSED [B,E,S]
//   4) Sc = Q*K^T (f32 per batch)   5) P = softmax(Sc/sqrt(S)) (bf16)
//   6) O = P*V (NT via V^T)         7) out = O*Wp^T+bp (f32)
// GEMM staging per 32-deep k-step (double buffered):
//   A tile 128x32 : TENSOR_LOAD_TO_LDS (TDM descriptor, wave 0, TENSORcnt),
//                   with D# padding -> 80B LDS rows (bank-conflict free).
//   B tile 128x32 : GLOBAL_LOAD_ASYNC_TO_LDS_B128 per lane (ASYNCcnt).
// Wave tile 32x64 -> 8 v_wmma_f32_16x16x32_bf16 per k-step.
// ---------------------------------------------------------------------------

typedef __attribute__((ext_vector_type(16))) __bf16 bf16x16;
typedef __attribute__((ext_vector_type(8)))  __bf16 bf16x8;
typedef __attribute__((ext_vector_type(8)))  float  f32x8;
typedef __attribute__((ext_vector_type(4)))  float  f32x4;
typedef __attribute__((ext_vector_type(4)))  unsigned u32x4;
typedef __attribute__((ext_vector_type(8)))  unsigned u32x8;

static __device__ __forceinline__ __bf16 f2bf(float f) { return (__bf16)f; }

// LDS byte offset of a generic pointer known to live in LDS.
static __device__ __forceinline__ unsigned lds_off_of(const void* p) {
    return (unsigned)(uintptr_t)(__attribute__((address_space(3))) const void*)p;
}

// gfx1250 async copy: 16 bytes per lane, global -> LDS, tracked by ASYNCcnt.
static __device__ __forceinline__ void async_copy_b128(unsigned lds_off, const void* gaddr) {
    asm volatile("global_load_async_to_lds_b128 %0, %1, off"
                 :: "v"(lds_off), "v"(gaddr) : "memory");
}
static __device__ __forceinline__ void wait_async0() {
    asm volatile("s_wait_asynccnt 0x0" ::: "memory");
}
static __device__ __forceinline__ void wait_tensor0() {
    __builtin_amdgcn_s_wait_tensorcnt(0);
}

// --- Tensor Data Mover descriptor (ISA 8.3/8.4), 2D tile, 2-byte elements ---
// Group 0: count=1 | lds_addr | global_addr | type=2.
static __device__ __forceinline__ u32x4 tdm_group0(unsigned lds_byte,
                                                   unsigned long long gaddr) {
    u32x4 g;
    g[0] = 1u;                                   // count=1 (valid user D#)
    g[1] = lds_byte;                             // lds_addr
    g[2] = (unsigned)gaddr;                      // global_addr[31:0]
    g[3] = ((unsigned)(gaddr >> 32) & 0x01FFFFFFu) | (2u << 30);  // [56:32] | type=2
    return g;
}
// Group 1: data_size=2B; pad_enable, pad every 16 DW (one 64B row) by 4 DW
// (16B) -> 80B LDS rows; tensor dims huge (no OOB clamp); tile 32 x 128;
// dim0 stride = row stride in elements.
static __device__ __forceinline__ u32x8 tdm_group1(unsigned stride_elems) {
    u32x8 g;
    g[0] = (1u << 16) | (1u << 20) | (3u << 22) | (3u << 25);
    g[1] = 0xFFFF0000u;                 // tensor_dim0[15:0]=0xFFFF (of 0x7FFFFFFF)
    g[2] = 0xFFFF7FFFu;                 // tensor_dim0[31:16] | tensor_dim1[15:0]
    g[3] = (32u << 16) | 0x7FFFu;       // tensor_dim1[31:16] | tile_dim0=32
    g[4] = 128u;                        // tile_dim1=128, tile_dim2=0
    g[5] = stride_elems;                // tensor_dim0_stride[31:0]
    g[6] = 0u;                          // stride[47:32] | dim1_stride[15:0]
    g[7] = 0u;
    return g;
}
static __device__ __forceinline__ void tdm_load(u32x4 g0, u32x8 g1) {
    asm volatile("tensor_load_to_lds %0, %1" :: "s"(g0), "s"(g1) : "memory");
}

// ---------------------------------------------------------------------------
// f32 -> bf16 conversion pass: 8 elements/thread, vector loads/stores.
// ---------------------------------------------------------------------------
__global__ __launch_bounds__(256)
void cvt_f32_to_bf16(const float* __restrict__ in, void* __restrict__ outv, long long n) {
    const size_t i = ((size_t)blockIdx.x * 256 + threadIdx.x) * 8;
    if ((long long)i >= n) return;
    f32x4 lo = *(const f32x4*)(in + i);
    f32x4 hi = *(const f32x4*)(in + i + 4);
    bf16x8 o;
#pragma unroll
    for (int j = 0; j < 4; ++j) { o[j] = f2bf(lo[j]); o[4 + j] = f2bf(hi[j]); }
    *((bf16x8*)outv + i / 8) = o;
}

// A fragment: 16x32 (MxK) bf16. Lanes 0-15: row m=lane, K={0..7,16..23};
// lanes 16-31: row m=lane-16, K={8..15,24..31}.  (ISA 7.12.2)
static __device__ __forceinline__ bf16x16 load_frag_a(const __bf16* p) {
    bf16x8 lo = *(const bf16x8*)p;
    bf16x8 hi = *(const bf16x8*)(p + 16);
    bf16x16 f;
#pragma unroll
    for (int i = 0; i < 8; ++i) { f[i] = lo[i]; f[8 + i] = hi[i]; }
    return f;
}

// B fragment: 32x16 (KxN) bf16 from row-major [N,K] tile (NT GEMM).
static __device__ __forceinline__ bf16x16 load_frag_b(const __bf16* p) {
    return load_frag_a(p);
}

enum StoreMode { STORE_BF16 = 0, STORE_F32_RAW = 1, STORE_F32_BIAS = 2, STORE_BF16_TRANS = 3 };

// C/D tile layout (ISA 7.12.2): element (m,n): n = lane&15, m = g + 8*(lane>>4).
template<int STORE>
static __device__ __forceinline__ void store_tile(void* C, const float* bias, f32x8 acc,
                                                  size_t row0, size_t col0, int ldc,
                                                  int lane, int S_len, int Ncols) {
    const int    n   = lane & 15;
    const int    mh  = (lane >> 4) * 8;
    const size_t col = col0 + n;
    float bv = 0.0f;
    if constexpr (STORE != STORE_F32_RAW) { if (bias) bv = bias[col]; }

    if constexpr (STORE == STORE_BF16) {
        __bf16* Cb = (__bf16*)C;
#pragma unroll
        for (int g = 0; g < 8; ++g)
            Cb[(row0 + mh + g) * (size_t)ldc + col] = f2bf(acc[g] + bv);
    } else if constexpr (STORE == STORE_F32_RAW) {
        float* Cf = (float*)C;
#pragma unroll
        for (int g = 0; g < 8; ++g)
            Cf[(row0 + mh + g) * (size_t)ldc + col] = acc[g];
    } else if constexpr (STORE == STORE_F32_BIAS) {
        float* Cf = (float*)C;
#pragma unroll
        for (int g = 0; g < 8; ++g)
            Cf[(row0 + mh + g) * (size_t)ldc + col] = acc[g] + bv;
    } else {  // STORE_BF16_TRANS: C is Vt[B][Ncols][S_len]; 8 contiguous bf16/lane
        const size_t batch = row0 / (size_t)S_len;
        const size_t s0    = row0 % (size_t)S_len + mh;
        __bf16* dst = (__bf16*)C + batch * (size_t)Ncols * S_len + col * (size_t)S_len + s0;
        bf16x8 v;
#pragma unroll
        for (int g = 0; g < 8; ++g) v[g] = f2bf(acc[g] + bv);
        *(bf16x8*)dst = v;
    }
}

// LDS tile row stride: 32 data + 8 pad bf16 => 80B rows; 20*r mod 64 is a
// permutation over 16 rows -> conflict-free ds_load_b128 fragment reads.
#define LDSS 40

// NT GEMM: C[M,N] = A[M,K]*B[N,K]^T (+bias). 256 threads = 8 waves.
// Block tile 128x128; A staged via TDM, B via per-lane async; double buffered.
template<int STORE>
__global__ __launch_bounds__(256)
void gemm_nt_wmma(const void* __restrict__ Av, const void* __restrict__ Bv,
                  void* __restrict__ C, const float* __restrict__ bias,
                  int M, int N, int K, int lda, int ldb, int ldc,
                  long long sAbytes, long long sBbytes, long long sCbytes,
                  int S_len) {
    __shared__ __align__(16) __bf16 As[2][128 * LDSS];
    __shared__ __align__(16) __bf16 Bs[2][128 * LDSS];

    const int tid  = threadIdx.x;
    const int lane = tid & 31;
    const int wave = tid >> 5;
    const size_t rowB = (size_t)blockIdx.x * 128;
    const size_t colB = (size_t)blockIdx.y * 128;
    const int wm = (wave & 3) * 32;   // wave M offset in block tile
    const int wn = (wave >> 2) * 64;  // wave N offset in block tile

    const __bf16* A = (const __bf16*)((const char*)Av + (size_t)blockIdx.z * (size_t)sAbytes);
    const __bf16* B = (const __bf16*)((const char*)Bv + (size_t)blockIdx.z * (size_t)sBbytes);
    void*        Cp = (char*)C + (size_t)blockIdx.z * (size_t)sCbytes;

    // TDM descriptors for the A tile (uniform scalars -> SGPR groups).
    const unsigned asBase[2] = { lds_off_of(&As[0][0]), lds_off_of(&As[1][0]) };
    const u32x8 g1A = tdm_group1((unsigned)lda);
    const __bf16* gAtile = A + rowB * (size_t)lda;

    // B tile async mapping: 128x32 = 512 chunks of 8 bf16; thread owns 2.
    const int c0 = tid * 2, c1 = c0 + 1;
    const int r0 = c0 >> 2, e0 = (c0 & 3) * 8;
    const int r1 = c1 >> 2, e1 = (c1 & 3) * 8;
    const __bf16* gB0 = B + (colB + r0) * (size_t)ldb + e0;
    const __bf16* gB1 = B + (colB + r1) * (size_t)ldb + e1;
    unsigned lB0[2], lB1[2];
#pragma unroll
    for (int b = 0; b < 2; ++b) {
        lB0[b] = lds_off_of(&Bs[b][r0 * LDSS + e0]);
        lB1[b] = lds_off_of(&Bs[b][r1 * LDSS + e1]);
    }

    // Prologue: stage k=0 tiles into buffer 0.
    if (wave == 0)
        tdm_load(tdm_group0(asBase[0], (unsigned long long)(uintptr_t)gAtile), g1A);
    async_copy_b128(lB0[0], gB0);
    async_copy_b128(lB1[0], gB1);
    wait_async0();
    wait_tensor0();
    __syncthreads();

    f32x8 acc[2][4] = {};
    int cur = 0;
    for (int k0 = 0; k0 < K; k0 += 32) {
        const int nxt = cur ^ 1;
        if (k0 + 32 < K) {   // stage next k-slice while computing this one
            if (wave == 0)
                tdm_load(tdm_group0(asBase[nxt],
                         (unsigned long long)(uintptr_t)(gAtile + k0 + 32)), g1A);
            async_copy_b128(lB0[nxt], gB0 + k0 + 32);
            async_copy_b128(lB1[nxt], gB1 + k0 + 32);
        }

        const __bf16* ap = &As[cur][(wm + (lane & 15)) * LDSS + (lane >> 4) * 8];
        const __bf16* bp = &Bs[cur][(wn + (lane & 15)) * LDSS + (lane >> 4) * 8];
        bf16x16 a0 = load_frag_a(ap);
        bf16x16 a1 = load_frag_a(ap + 16 * LDSS);
        bf16x16 b0 = load_frag_b(bp);
        bf16x16 b1 = load_frag_b(bp + 16 * LDSS);
        bf16x16 b2 = load_frag_b(bp + 32 * LDSS);
        bf16x16 b3 = load_frag_b(bp + 48 * LDSS);
        acc[0][0] = __builtin_amdgcn_wmma_f32_16x16x32_bf16(false, a0, false, b0, (short)0, acc[0][0], false, false);
        acc[0][1] = __builtin_amdgcn_wmma_f32_16x16x32_bf16(false, a0, false, b1, (short)0, acc[0][1], false, false);
        acc[0][2] = __builtin_amdgcn_wmma_f32_16x16x32_bf16(false, a0, false, b2, (short)0, acc[0][2], false, false);
        acc[0][3] = __builtin_amdgcn_wmma_f32_16x16x32_bf16(false, a0, false, b3, (short)0, acc[0][3], false, false);
        acc[1][0] = __builtin_amdgcn_wmma_f32_16x16x32_bf16(false, a1, false, b0, (short)0, acc[1][0], false, false);
        acc[1][1] = __builtin_amdgcn_wmma_f32_16x16x32_bf16(false, a1, false, b1, (short)0, acc[1][1], false, false);
        acc[1][2] = __builtin_amdgcn_wmma_f32_16x16x32_bf16(false, a1, false, b2, (short)0, acc[1][2], false, false);
        acc[1][3] = __builtin_amdgcn_wmma_f32_16x16x32_bf16(false, a1, false, b3, (short)0, acc[1][3], false, false);

        wait_async0();       // this wave's B copies into `nxt` have landed
        wait_tensor0();      // wave 0: A TDM into `nxt` done (no-op elsewhere)
        __syncthreads();     // everyone done reading `cur` / writing `nxt`
        cur = nxt;
    }

#pragma unroll
    for (int i = 0; i < 2; ++i)
#pragma unroll
        for (int j = 0; j < 4; ++j)
            store_tile<STORE>(Cp, bias, acc[i][j], rowB + wm + i * 16, colB + wn + j * 16,
                              ldc, lane, S_len, N);
}

// Row softmax: one 256-thread block per row of 2048 f32 scores -> bf16 P.
__global__ __launch_bounds__(256)
void softmax_rows(const float* __restrict__ Sc, void* __restrict__ Pv, int S, float scale) {
    const int row = blockIdx.x;
    const float* x = Sc + (size_t)row * S;
    __bf16* p = (__bf16*)Pv + (size_t)row * S;
    const int tid = threadIdx.x;

    float v[8];
    float lmax = -3.4e38f;
#pragma unroll
    for (int i = 0; i < 8; ++i) {
        v[i] = x[tid + 256 * i] * scale;
        lmax = fmaxf(lmax, v[i]);
    }
    __shared__ float red[256];
    red[tid] = lmax; __syncthreads();
    for (int off = 128; off > 0; off >>= 1) {
        if (tid < off) red[tid] = fmaxf(red[tid], red[tid + off]);
        __syncthreads();
    }
    const float m = red[0];
    __syncthreads();

    float lsum = 0.f;
#pragma unroll
    for (int i = 0; i < 8; ++i) { v[i] = __expf(v[i] - m); lsum += v[i]; }
    red[tid] = lsum; __syncthreads();
    for (int off = 128; off > 0; off >>= 1) {
        if (tid < off) red[tid] += red[tid + off];
        __syncthreads();
    }
    const float inv = 1.0f / red[0];
#pragma unroll
    for (int i = 0; i < 8; ++i) p[tid + 256 * i] = f2bf(v[i] * inv);
}

extern "C" void kernel_launch(void* const* d_in, const int* in_sizes, int n_in,
                              void* d_out, int out_size, void* d_ws, size_t ws_size,
                              hipStream_t stream) {
    (void)in_sizes; (void)n_in; (void)out_size; (void)ws_size;
    const float* x  = (const float*)d_in[0];
    const float* Wq = (const float*)d_in[1];
    const float* bq = (const float*)d_in[2];
    const float* Wk = (const float*)d_in[3];
    const float* bk = (const float*)d_in[4];
    const float* Wv = (const float*)d_in[5];
    const float* bv = (const float*)d_in[6];
    const float* Wp = (const float*)d_in[7];
    const float* bp = (const float*)d_in[8];
    float* out = (float*)d_out;

    const int Bz = 4, S = 2048, E = 1024;
    const int M = Bz * S;  // 8192

    // Workspace carve-out (~192 MB total)
    char* w = (char*)d_ws;
    void* xb  = (void*)w; w += (size_t)M * E * 2;          // bf16 [B*S, E]
    void* Wqb = (void*)w; w += (size_t)E * E * 2;          // bf16 [E, E]
    void* Wkb = (void*)w; w += (size_t)E * E * 2;
    void* Wvb = (void*)w; w += (size_t)E * E * 2;
    void* Wpb = (void*)w; w += (size_t)E * E * 2;
    void* Qb  = (void*)w; w += (size_t)M * E * 2;          // bf16 [B*S, E]
    void* Kb  = (void*)w; w += (size_t)M * E * 2;          // bf16 [B*S, E]
    void* Vt  = (void*)w; w += (size_t)M * E * 2;          // bf16 [B, E, S]
    float* Sc = (float*)w; w += (size_t)Bz * S * S * 4;    // f32  [B, S, S]
    void* P   = (void*)w; w += (size_t)Bz * S * S * 2;     // bf16 [B, S, S]
    void* Ob  = (void*)w; w += (size_t)M * E * 2;          // bf16 [B*S, E]

    const dim3 blk(256);

    // 0) one-time conversions
    const long long nx = (long long)M * E, nw = (long long)E * E;
    cvt_f32_to_bf16<<<dim3((unsigned)(nx / 8 / 256)), blk, 0, stream>>>(x,  xb,  nx);
    cvt_f32_to_bf16<<<dim3((unsigned)(nw / 8 / 256)), blk, 0, stream>>>(Wq, Wqb, nw);
    cvt_f32_to_bf16<<<dim3((unsigned)(nw / 8 / 256)), blk, 0, stream>>>(Wk, Wkb, nw);
    cvt_f32_to_bf16<<<dim3((unsigned)(nw / 8 / 256)), blk, 0, stream>>>(Wv, Wvb, nw);
    cvt_f32_to_bf16<<<dim3((unsigned)(nw / 8 / 256)), blk, 0, stream>>>(Wp, Wpb, nw);

    // 1-3) QKV projections
    const dim3 g1(M / 128, E / 128, 1);
    gemm_nt_wmma<STORE_BF16><<<g1, blk, 0, stream>>>(
        xb, Wqb, Qb, bq, M, E, E, E, E, E, 0, 0, 0, S);
    gemm_nt_wmma<STORE_BF16><<<g1, blk, 0, stream>>>(
        xb, Wkb, Kb, bk, M, E, E, E, E, E, 0, 0, 0, S);
    gemm_nt_wmma<STORE_BF16_TRANS><<<g1, blk, 0, stream>>>(
        xb, Wvb, Vt, bv, M, E, E, E, E, S, 0, 0, 0, S);

    // 4) scores = Q * K^T per batch (f32 out; scale folded into softmax)
    const dim3 g2(S / 128, S / 128, Bz);
    gemm_nt_wmma<STORE_F32_RAW><<<g2, blk, 0, stream>>>(
        Qb, Kb, Sc, nullptr, S, S, E, E, E, S,
        (long long)S * E * 2, (long long)S * E * 2, (long long)S * S * 4, S);

    // 5) softmax rows (replicates reference's 1/sqrt(seq_len) scaling)
    const float scale = 1.0f / sqrtf((float)S);
    softmax_rows<<<dim3((unsigned)M), blk, 0, stream>>>(Sc, P, S, scale);

    // 6) O = P * V  (NT via transposed V: B = Vt[E,S], ldb = S)
    const dim3 g3(S / 128, E / 128, Bz);
    gemm_nt_wmma<STORE_BF16><<<g3, blk, 0, stream>>>(
        P, Vt, Ob, nullptr, S, E, S, S, S, E,
        (long long)S * S * 2, (long long)E * S * 2, (long long)S * E * 2, S);

    // 7) out = O * Wp^T + bp (f32 out)
    gemm_nt_wmma<STORE_F32_BIAS><<<g1, blk, 0, stream>>>(
        Ob, Wpb, out, bp, M, E, E, E, E, E, 0, 0, 0, S);
}
